// Model_Contrast_47974784697102
// MI455X (gfx1250) — matrix operand order repaired and checked by reading, not compiled, not run on hardware
//
#include <hip/hip_runtime.h>
#include <hip/hip_bf16.h>

typedef __attribute__((ext_vector_type(16))) _Float16 v16h;
typedef __attribute__((ext_vector_type(8)))  _Float16 v8h;
typedef __attribute__((ext_vector_type(4)))  _Float16 v4h;
typedef __attribute__((ext_vector_type(8)))  float    v8f;

#define HIDDEN 256
#define NROWS  8192
#define TAU_INV 2.0f   // 1 / 0.5

// ---- workspace layout (bytes) ----
// acc[2]  : global P / Q accumulators
// xh      : f16 copy of [v1 ; v2]  (16384 x 256)
// Wh      : f16 copy of W          (256 x 256)
// zh      : f16 [z1 ; z2]          (16384 x 256)
// rn      : f32 reciprocal norms   (16384)   [0:8192)=rn1, [8192:)=rn2
static const size_t WS_ACC = 0;
static const size_t WS_XH  = 256;
static const size_t WS_WH  = WS_XH + (size_t)16384 * 256 * 2;
static const size_t WS_ZH  = WS_WH + (size_t)256 * 256 * 2;
static const size_t WS_RN  = WS_ZH + (size_t)16384 * 256 * 2;

// ---------------- init ----------------
__global__ void k_init(float* acc) {
    if (threadIdx.x < 2) acc[threadIdx.x] = 0.0f;
}

// ---------------- f32 -> f16 convert (vector of 4) ----------------
__global__ void k_cvt(const float* __restrict__ in, _Float16* __restrict__ out, int n4) {
    int i = blockIdx.x * blockDim.x + threadIdx.x;
    if (i >= n4) return;
    float4 v = ((const float4*)in)[i];
    v4h h;
    h[0] = (_Float16)v.x; h[1] = (_Float16)v.y;
    h[2] = (_Float16)v.z; h[3] = (_Float16)v.w;
    ((v4h*)out)[i] = h;
}

// Build an A-fragment (16x32 f16) per the CDNA5 VGPR layout:
// lanes 0-15: K = k0+0..7 (VGPR0-3), k0+16..23 (VGPR4-7)
// lanes16-31: K = k0+8..15,          k0+24..31
__device__ __forceinline__ v16h load_a_frag(const _Float16* row, int k0, int sel) {
    union { v16h v; v8h h[2]; } u;
    u.h[0] = *(const v8h*)(row + k0 + sel * 8);
    u.h[1] = *(const v8h*)(row + k0 + 16 + sel * 8);
    return u.v;
}

// Build a B-fragment (32x16 f16, column-major source = row-major z/W rows):
// lanes 0-15 hold K = k0+0..7, k0+16..23 of column (lane), lanes 16-31 hold K = k0+8..15, k0+24..31
__device__ __forceinline__ v16h load_b_frag(const _Float16* row, int k0, int sel) {
    union { v16h v; v8h h[2]; } u;
    u.h[0] = *(const v8h*)(row + k0 + sel * 8);
    u.h[1] = *(const v8h*)(row + k0 + 16 + sel * 8);
    return u.v;
}

// ---------------- projection: z = elu(x @ W^T + b), rn = 1/||z_row|| ----------------
// block: 256 threads = 8 waves; wave handles 16 rows (one M tile), all 256 cols.
__global__ __launch_bounds__(256) void k_proj(const _Float16* __restrict__ xh,
                                              const _Float16* __restrict__ Wh,
                                              const float* __restrict__ bias,
                                              _Float16* __restrict__ zh,
                                              float* __restrict__ rn) {
    const int lane = threadIdx.x & 31;
    const int wave = threadIdx.x >> 5;
    const int sel  = lane >> 4;
    const int l15  = lane & 15;
    const int Rbase = blockIdx.x * 128 + wave * 16;

    // cache all 8 A fragments (K = 256) for this wave's 16 rows
    const _Float16* arow = xh + (size_t)(Rbase + l15) * HIDDEN;
    v16h a[8];
#pragma unroll
    for (int kk = 0; kk < 8; ++kk) a[kk] = load_a_frag(arow, kk * 32, sel);

    float sq[8];
#pragma unroll
    for (int r = 0; r < 8; ++r) sq[r] = 0.0f;

    for (int nt = 0; nt < 16; ++nt) {
        const _Float16* brow = Wh + (size_t)(nt * 16 + l15) * HIDDEN;
        v8f c = {};
#pragma unroll
        for (int kk = 0; kk < 8; ++kk) {
            v16h bfr = load_b_frag(brow, kk * 32, sel);
            c = __builtin_amdgcn_wmma_f32_16x16x32_f16(false, a[kk], false, bfr,
                                                       (short)0, c, false, false);
        }
        const int col = nt * 16 + l15;
        const float bb = bias[col];
#pragma unroll
        for (int r = 0; r < 8; ++r) {
            float v = c[r] + bb;
            v = (v > 0.0f) ? v : expm1f(v);   // ELU, alpha = 1
            sq[r] += v * v;
            const int row = Rbase + r + 8 * sel;
            zh[(size_t)row * HIDDEN + col] = (_Float16)v;
        }
    }
    // reduce sum-of-squares across the 16-lane half that shares each row
#pragma unroll
    for (int r = 0; r < 8; ++r) {
        float s = sq[r];
        s += __shfl_xor(s, 1, 16);
        s += __shfl_xor(s, 2, 16);
        s += __shfl_xor(s, 4, 16);
        s += __shfl_xor(s, 8, 16);
        if (l15 == 0) rn[Rbase + r + 8 * sel] = rsqrtf(s);
    }
}

// ---------------- main: S = z1 @ z2^T tile, fused exp/weight/reduce ----------------
// block: 256 threads = 8 waves, covers a 128x128 output tile.
// wave (w>>2) in {0,1} picks 64-row half; (w&3) picks 32-col quarter.
__global__ __launch_bounds__(256) void k_main(const _Float16* __restrict__ z1h,
                                              const _Float16* __restrict__ z2h,
                                              const float* __restrict__ rn1,
                                              const float* __restrict__ rn2,
                                              const float* __restrict__ pos,
                                              const float* __restrict__ neg,
                                              float* __restrict__ acc) {
    const int lane = threadIdx.x & 31;
    const int wave = threadIdx.x >> 5;
    const int sel  = lane >> 4;
    const int l15  = lane & 15;
    const int bx = blockIdx.x & 63;
    const int by = blockIdx.x >> 6;
    const int Rbase = by * 128 + (wave >> 2) * 64;   // 4 M tiles
    const int Cbase = bx * 128 + (wave & 3) * 32;    // 2 N tiles

    v8f c[4][2];
#pragma unroll
    for (int mt = 0; mt < 4; ++mt)
#pragma unroll
        for (int nt = 0; nt < 2; ++nt) c[mt][nt] = (v8f){};

    for (int kk = 0; kk < 8; ++kk) {
        const int k0 = kk * 32;
        v16h a[4], b[2];
#pragma unroll
        for (int mt = 0; mt < 4; ++mt)
            a[mt] = load_a_frag(z1h + (size_t)(Rbase + mt * 16 + l15) * HIDDEN, k0, sel);
#pragma unroll
        for (int nt = 0; nt < 2; ++nt)
            b[nt] = load_b_frag(z2h + (size_t)(Cbase + nt * 16 + l15) * HIDDEN, k0, sel);
#pragma unroll
        for (int mt = 0; mt < 4; ++mt)
#pragma unroll
            for (int nt = 0; nt < 2; ++nt)
                c[mt][nt] = __builtin_amdgcn_wmma_f32_16x16x32_f16(
                    false, a[mt], false, b[nt], (short)0, c[mt][nt], false, false);
    }

    float sp = 0.0f, sn = 0.0f;
#pragma unroll
    for (int mt = 0; mt < 4; ++mt) {
        float r1v[8];
#pragma unroll
        for (int r = 0; r < 8; ++r) r1v[r] = rn1[Rbase + mt * 16 + r + 8 * sel];
#pragma unroll
        for (int nt = 0; nt < 2; ++nt) {
            const int col = Cbase + nt * 16 + l15;
            const float r2 = rn2[col];
#pragma unroll
            for (int r = 0; r < 8; ++r) {
                const int row = Rbase + mt * 16 + r + 8 * sel;
                const float cosv = c[mt][nt][r] * r1v[r] * r2;
                const float e = __expf(cosv * TAU_INV);
                const size_t idx = (size_t)row * NROWS + col;
                const float p = __builtin_nontemporal_load(pos + idx);
                const float q = __builtin_nontemporal_load(neg + idx);
                sp = fmaf(e, p, sp);
                sn = fmaf(e, q, sn);
            }
        }
    }

    // wave reduce (wave32)
    for (int m = 1; m < 32; m <<= 1) {
        sp += __shfl_xor(sp, m, 32);
        sn += __shfl_xor(sn, m, 32);
    }
    __shared__ float sP[8], sN[8];
    if (lane == 0) { sP[wave] = sp; sN[wave] = sn; }
    __syncthreads();
    if (threadIdx.x == 0) {
        float tp = 0.0f, tn = 0.0f;
#pragma unroll
        for (int w = 0; w < 8; ++w) { tp += sP[w]; tn += sN[w]; }
        atomicAdd(&acc[0], tp);
        atomicAdd(&acc[1], tn);
    }
}

// ---------------- final: -log(P / (P + Q)) ----------------
__global__ void k_final(const float* __restrict__ acc, float* __restrict__ out) {
    const float P = acc[0], Q = acc[1];
    out[0] = -logf(P / (P + Q));
}

extern "C" void kernel_launch(void* const* d_in, const int* in_sizes, int n_in,
                              void* d_out, int out_size, void* d_ws, size_t ws_size,
                              hipStream_t stream) {
    (void)in_sizes; (void)n_in; (void)out_size; (void)ws_size;
    const float* v1  = (const float*)d_in[0];
    const float* v2  = (const float*)d_in[1];
    const float* pos = (const float*)d_in[2];
    const float* neg = (const float*)d_in[3];
    const float* W   = (const float*)d_in[4];
    const float* b   = (const float*)d_in[5];
    float* out = (float*)d_out;

    char* ws = (char*)d_ws;
    float*    acc = (float*)(ws + WS_ACC);
    _Float16* xh  = (_Float16*)(ws + WS_XH);
    _Float16* Wh  = (_Float16*)(ws + WS_WH);
    _Float16* zh  = (_Float16*)(ws + WS_ZH);
    float*    rn  = (float*)(ws + WS_RN);

    k_init<<<1, 32, 0, stream>>>(acc);

    const int n4 = NROWS * HIDDEN / 4;
    k_cvt<<<(n4 + 255) / 256, 256, 0, stream>>>(v1, xh, n4);
    k_cvt<<<(n4 + 255) / 256, 256, 0, stream>>>(v2, xh + (size_t)NROWS * HIDDEN, n4);
    const int w4 = HIDDEN * HIDDEN / 4;
    k_cvt<<<(w4 + 255) / 256, 256, 0, stream>>>(W, Wh, w4);

    // 16384 rows / 128 rows-per-block
    k_proj<<<128, 256, 0, stream>>>(xh, Wh, b, zh, rn);

    // 64 x 64 blocks of 128x128 output tiles
    k_main<<<64 * 64, 256, 0, stream>>>(zh, zh + (size_t)NROWS * HIDDEN,
                                        rn, rn + NROWS, pos, neg, acc);

    k_final<<<1, 1, 0, stream>>>(acc, out);
}
